// MultiheadSelfAttention_30958124269835
// MI455X (gfx1250) — matrix operand order repaired and checked by reading, not compiled, not run on hardware
//
#include <hip/hip_runtime.h>
#include <hip/hip_bf16.h>

#define Bsz 2
#define Ssz 2048
#define Dsz 1024
#define Hn  16
#define DKs 64
#define Mrows (Bsz*Ssz)   // 4096

typedef _Float16 half8_t __attribute__((ext_vector_type(8)));
typedef _Float16 v16h    __attribute__((ext_vector_type(16)));
typedef float    v8f     __attribute__((ext_vector_type(8)));

union AFrag { v16h v; half8_t h[2]; };

static __device__ __forceinline__ v8f vzero8() {
  v8f v;
#pragma unroll
  for (int i = 0; i < 8; i++) v[i] = 0.0f;
  return v;
}

static __device__ __forceinline__ v8f wmma_f16(v16h a, v16h b, v8f c) {
  // D = A(16x32 f16) * B(32x16 f16) + C(16x16 f32)
  return __builtin_amdgcn_wmma_f32_16x16x32_f16(false, a, false, b, (short)0, c, false, false);
}

__global__ void cvt_f32_to_f16(const float* __restrict__ in, _Float16* __restrict__ out, int n) {
  int i = blockIdx.x * blockDim.x + threadIdx.x;
  if (i < n) out[i] = (_Float16)in[i];
}

// C[M x N] = A[M x K] * W^T  (W is [N x K] row-major, i.e. torch Linear weight)
// mode 0: write f16 to [b][h][s][dk]   (Q / K layout)
// mode 1: write f16 to [b][h][dk][s]   (V transposed)
// mode 2: write f32 row-major M x N    (final output)
__global__ __launch_bounds__(128)
void gemm_wmma(const _Float16* __restrict__ A, const _Float16* __restrict__ W,
               void* __restrict__ Cout, int K, int N, int mode)
{
  const int lane = threadIdx.x & 31;
  const int wave = threadIdx.x >> 5;
  const int hlf  = lane >> 4;
  const int l    = lane & 15;
  const int wm = wave >> 1, wn = wave & 1;      // 2x2 waves -> 128x128 block tile
  const int m0 = blockIdx.x * 128 + wm * 64;
  const int n0 = blockIdx.y * 128 + wn * 64;

  v8f c[4][4];
#pragma unroll
  for (int i = 0; i < 4; i++)
#pragma unroll
    for (int j = 0; j < 4; j++) c[i][j] = vzero8();

  for (int k = 0; k < K; k += 32) {
    AFrag a[4]; v16h b[4];
#pragma unroll
    for (int mi = 0; mi < 4; mi++) {
      const _Float16* p = A + (size_t)(m0 + mi * 16 + l) * K + k + hlf * 16;
      a[mi].h[0] = *(const half8_t*)p;
      a[mi].h[1] = *(const half8_t*)(p + 8);
    }
#pragma unroll
    for (int ni = 0; ni < 4; ni++) {
      const _Float16* p = W + (size_t)(n0 + ni * 16 + l) * K + k + hlf * 16;
      b[ni] = *(const v16h*)p;
    }
#pragma unroll
    for (int mi = 0; mi < 4; mi++)
#pragma unroll
      for (int ni = 0; ni < 4; ni++)
        c[mi][ni] = wmma_f16(a[mi].v, b[ni], c[mi][ni]);
  }

#pragma unroll
  for (int mi = 0; mi < 4; mi++)
#pragma unroll
    for (int ni = 0; ni < 4; ni++)
#pragma unroll
      for (int r = 0; r < 8; r++) {
        int m = m0 + mi * 16 + hlf * 8 + r;
        int n = n0 + ni * 16 + l;
        float v = c[mi][ni][r];
        if (mode == 2) {
          ((float*)Cout)[(size_t)m * N + n] = v;
        } else {
          int b_ = m >> 11, s = m & (Ssz - 1);
          int h_ = n >> 6,  d = n & (DKs - 1);
          size_t idx = (mode == 0)
              ? ((size_t)(b_ * Hn + h_) * Ssz + s) * DKs + d
              : ((size_t)(b_ * Hn + h_) * DKs + d) * Ssz + s;
          ((_Float16*)Cout)[idx] = (_Float16)v;
        }
      }
}

// Flash-style causal attention.
// 8 waves per workgroup, all on the SAME (b,h), covering 128 consecutive queries:
// the waves issue identical K/V cacheline addresses so the WGP$ captures the reuse.
// Each wave owns a private LDS region; ordering is wave-local (LDS is in-order per
// wave on CDNA5) -> s_wait_dscnt + compiler fence instead of block barriers.
// Q,K: [b][h][s][dk] f16 ; Vt: [b][h][dk][s] f16 ; ctx out: [b][s][h*dk] f16
__global__ __launch_bounds__(256)
void attn_wmma(const _Float16* __restrict__ Qm, const _Float16* __restrict__ Km,
               const _Float16* __restrict__ Vt, _Float16* __restrict__ ctx)
{
  __shared__ __align__(16) _Float16 ldsP[8][16][40];   // per-wave P transpose staging
  const int tid  = threadIdx.x;
  const int wave = tid >> 5;
  const int lane = tid & 31;
  const int hlf  = lane >> 4, l = lane & 15;
  const int bh   = blockIdx.y;
  const int b_   = bh >> 4, h_ = bh & 15;
  const int qbase = (blockIdx.x * 8 + wave) * 16;
  const _Float16* Qh = Qm + (size_t)bh * Ssz * DKs;
  const _Float16* Kh = Km + (size_t)bh * Ssz * DKs;
  const _Float16* Vh = Vt + (size_t)bh * DKs * Ssz;
  _Float16 (*myP)[40] = ldsP[wave];

  AFrag qf[2];
#pragma unroll
  for (int kk = 0; kk < 2; kk++) {
    const _Float16* p = Qh + (size_t)(qbase + l) * DKs + kk * 32 + hlf * 16;
    qf[kk].h[0] = *(const half8_t*)p;
    qf[kk].h[1] = *(const half8_t*)(p + 8);
  }

  v8f o[4];
#pragma unroll
  for (int f = 0; f < 4; f++) o[f] = vzero8();
  float mrun[8], lrun[8];
#pragma unroll
  for (int r = 0; r < 8; r++) { mrun[r] = -1e30f; lrun[r] = 0.0f; }

  for (int j = 0; j < qbase + 16; j += 32) {   // causal: only chunks with keys <= max query
    v8f s0 = vzero8(), s1 = vzero8();
#pragma unroll
    for (int kk = 0; kk < 2; kk++) {           // dk = 64 -> two K-steps of 32
      v16h k0 = *(const v16h*)(Kh + (size_t)(j + l)      * DKs + kk * 32 + hlf * 16);
      v16h k1 = *(const v16h*)(Kh + (size_t)(j + 16 + l) * DKs + kk * 32 + hlf * 16);
      s0 = wmma_f16(qf[kk].v, k0, s0);
      s1 = wmma_f16(qf[kk].v, k1, s1);
    }
    const int key0 = j + l, key1 = j + 16 + l;
#pragma unroll
    for (int r = 0; r < 8; r++) {
      int q = qbase + hlf * 8 + r;
      float v0 = (key0 <= q) ? s0[r] * 0.125f : -1e30f;   // 1/sqrt(64)=0.125
      float v1 = (key1 <= q) ? s1[r] * 0.125f : -1e30f;
      float mx = fmaxf(v0, v1);
#pragma unroll
      for (int d = 1; d < 16; d <<= 1) mx = fmaxf(mx, __shfl_xor(mx, d, 32));
      float mnew = fmaxf(mrun[r], mx);
      float sc = __expf(mrun[r] - mnew);
      float e0 = __expf(v0 - mnew);
      float e1 = __expf(v1 - mnew);
      float rs = e0 + e1;
#pragma unroll
      for (int d = 1; d < 16; d <<= 1) rs += __shfl_xor(rs, d, 32);
      lrun[r] = lrun[r] * sc + rs;
      mrun[r] = mnew;
#pragma unroll
      for (int f = 0; f < 4; f++) o[f][r] *= sc;
      myP[hlf * 8 + r][l]      = (_Float16)e0;   // C-layout -> [q][key] in LDS
      myP[hlf * 8 + r][16 + l] = (_Float16)e1;
    }
    // wave-local: make the 16x32 P tile visible before re-reading transposed.
    // LDS is in-order per wave; this is a counter-wait + compiler fence.
    asm volatile("s_wait_dscnt 0x0" ::: "memory");
    AFrag pf;                                     // reload as A-fragment (16x32)
    pf.h[0] = *(const half8_t*)&myP[l][hlf * 16];
    pf.h[1] = *(const half8_t*)&myP[l][hlf * 16 + 8];
    asm volatile("" ::: "memory");                // keep reads before next iter's writes
#pragma unroll
    for (int f = 0; f < 4; f++) {
      v16h vf = *(const v16h*)(Vh + (size_t)(f * 16 + l) * Ssz + j + hlf * 16);
      o[f] = wmma_f16(pf.v, vf, o[f]);
    }
  }

  float inv[8];
#pragma unroll
  for (int r = 0; r < 8; r++) inv[r] = 1.0f / lrun[r];
#pragma unroll
  for (int f = 0; f < 4; f++)
#pragma unroll
    for (int r = 0; r < 8; r++) {
      int q = qbase + hlf * 8 + r;
      int dcol = h_ * DKs + f * 16 + l;
      ctx[(size_t)(b_ * Ssz + q) * Dsz + dcol] = (_Float16)(o[f][r] * inv[r]);
    }
}

extern "C" void kernel_launch(void* const* d_in, const int* in_sizes, int n_in,
                              void* d_out, int out_size, void* d_ws, size_t ws_size,
                              hipStream_t stream) {
  const float* x  = (const float*)d_in[0];
  const float* Wq = (const float*)d_in[1];
  const float* Wk = (const float*)d_in[2];
  const float* Wv = (const float*)d_in[3];
  const float* Wo = (const float*)d_in[4];
  float* out = (float*)d_out;

  char* ws = (char*)d_ws;
  const size_t actSz = (size_t)Mrows * Dsz * 2;   // 8 MiB each
  const size_t wSz   = (size_t)Dsz * Dsz * 2;     // 2 MiB each
  _Float16* xh   = (_Float16*)(ws);
  _Float16* Wqh  = (_Float16*)(ws + actSz);
  _Float16* Wkh  = (_Float16*)(ws + actSz + wSz);
  _Float16* Wvh  = (_Float16*)(ws + actSz + 2 * wSz);
  _Float16* Woh  = (_Float16*)(ws + actSz + 3 * wSz);
  _Float16* Qh   = (_Float16*)(ws + actSz + 4 * wSz);
  _Float16* Kh   = (_Float16*)(ws + 2 * actSz + 4 * wSz);
  _Float16* Vth  = (_Float16*)(ws + 3 * actSz + 4 * wSz);
  _Float16* ctxh = (_Float16*)(ws + 4 * actSz + 4 * wSz);

  const int nx = Mrows * Dsz;       // 4M elems
  const int nw = Dsz * Dsz;         // 1M elems
  cvt_f32_to_f16<<<(nx + 255) / 256, 256, 0, stream>>>(x, xh, nx);
  cvt_f32_to_f16<<<(nw + 255) / 256, 256, 0, stream>>>(Wq, Wqh, nw);
  cvt_f32_to_f16<<<(nw + 255) / 256, 256, 0, stream>>>(Wk, Wkh, nw);
  cvt_f32_to_f16<<<(nw + 255) / 256, 256, 0, stream>>>(Wv, Wvh, nw);
  cvt_f32_to_f16<<<(nw + 255) / 256, 256, 0, stream>>>(Wo, Woh, nw);

  dim3 g(Mrows / 128, Dsz / 128);   // (32, 8)
  gemm_wmma<<<g, 128, 0, stream>>>(xh, Wqh, Qh,  Dsz, Dsz, 0);
  gemm_wmma<<<g, 128, 0, stream>>>(xh, Wkh, Kh,  Dsz, Dsz, 0);
  gemm_wmma<<<g, 128, 0, stream>>>(xh, Wvh, Vth, Dsz, Dsz, 1);

  // 8 waves/block = 128 consecutive queries of one head -> WGP$ captures K/V reuse
  attn_wmma<<<dim3(Ssz / 128, Bsz * Hn), 256, 0, stream>>>(Qh, Kh, Vth, ctxh);

  gemm_wmma<<<g, 128, 0, stream>>>(ctxh, Woh, out, Dsz, Dsz, 2);
}
